// OuterProductMean_4569845203032
// MI455X (gfx1250) — matrix operand order repaired and checked by reading, not compiled, not run on hardware
//
#include <hip/hip_runtime.h>

typedef __attribute__((ext_vector_type(16))) _Float16 v16h;
typedef __attribute__((ext_vector_type(8)))  float    v8f;
typedef __attribute__((ext_vector_type(2)))  _Float16 h2;

#define SEQ  192
#define BATCH 2
#define CP   128
#define C    32
#define CI   128
#define NCOL (CI * C)   // 4096

// ---------------- K1: LayerNorm + dual projection ----------------
// One block (128 threads) per (s,b) row. Produces f16 p1[b,s,c], p2[b,s,c].
__global__ void ln_proj_kernel(const float* __restrict__ p,
                               const float* __restrict__ gamma,
                               const float* __restrict__ beta,
                               const float* __restrict__ W1,
                               const float* __restrict__ b1,
                               const float* __restrict__ W2,
                               const float* __restrict__ b2,
                               _Float16* __restrict__ p1h,
                               _Float16* __restrict__ p2h) {
  __shared__ float red[128];
  __shared__ float pn[128];
  const int row = blockIdx.x;            // s*BATCH + b
  const int s = row / BATCH, b = row % BATCH;
  const int t = threadIdx.x;
  const float x = p[(size_t)s * BATCH * CP + (size_t)b * CP + t];

  red[t] = x; __syncthreads();
  for (int off = 64; off > 0; off >>= 1) {
    if (t < off) red[t] += red[t + off];
    __syncthreads();
  }
  const float mu = red[0] * (1.0f / 128.0f);
  __syncthreads();

  const float xm = x - mu;
  red[t] = xm * xm; __syncthreads();
  for (int off = 64; off > 0; off >>= 1) {
    if (t < off) red[t] += red[t + off];
    __syncthreads();
  }
  const float var = red[0] * (1.0f / 128.0f);
  const float inv = rsqrtf(var + 1e-5f);
  pn[t] = xm * inv * gamma[t] + beta[t];
  __syncthreads();

  if (t < 64) {
    const int c = t & 31;
    const float* W = (t < 32) ? W1 : W2;
    float acc = (t < 32) ? b1[c] : b2[c];
    #pragma unroll 4
    for (int q = 0; q < CP; ++q) acc += pn[q] * W[c * CP + q];
    _Float16* dst = (t < 32) ? p1h : p2h;
    dst[((size_t)b * SEQ + s) * C + c] = (_Float16)acc;
  }
}

// ---------------- K2: repack Wout -> Whc[(k*C+d)*C + c] (f16) ----------------
__global__ void wprep_kernel(const float* __restrict__ Wout,
                             _Float16* __restrict__ Whc) {
  const int idx = blockIdx.x * blockDim.x + threadIdx.x;  // over CI*C*C
  if (idx >= CI * C * C) return;
  const int k = idx / (C * C);
  const int r = idx % (C * C);
  const int c = r / C, d = r % C;
  Whc[((size_t)k * C + d) * C + c] = (_Float16)Wout[idx];
}

// ---------------- K3: stage-2 GEMM: T[b,i,col] = p1[b] (192x32) @ W' (32x4096)
// One wave per 16x16 tile, K=32 == one v_wmma_f32_16x16x32_f16.
__global__ void stage2_kernel(const _Float16* __restrict__ p1h,
                              const _Float16* __restrict__ Whc,
                              _Float16* __restrict__ Th) {
  const int wave = (int)((blockIdx.x * blockDim.x + threadIdx.x) >> 5);
  const int lane = threadIdx.x & 31;
  const int b   = wave / (12 * 256);
  const int rem = wave % (12 * 256);
  const int i0   = (rem / 256) * 16;
  const int col0 = (rem % 256) * 16;
  const int n = lane & 15, h = lane >> 4;

  // A: 16x32 f16 rows of p1
  const _Float16* Arow = p1h + ((size_t)b * SEQ + (i0 + n)) * C;
  v16h a, bm;
  #pragma unroll
  for (int v = 0; v < 8; ++v) {
    const int kb = h * 16 + 2 * v;
    h2 pa = *(const h2*)(Arow + kb);
    a[2 * v] = pa.x; a[2 * v + 1] = pa.y;
  }
  // B: 32x16; lane n holds column col0+n; contiguous in c
  const _Float16* Brow = Whc + (size_t)(col0 + n) * C + h * 16;
  #pragma unroll
  for (int v = 0; v < 8; ++v) {
    h2 pb = *(const h2*)(Brow + 2 * v);
    bm[2 * v] = pb.x; bm[2 * v + 1] = pb.y;
  }

  v8f acc = {};
  acc = __builtin_amdgcn_wmma_f32_16x16x32_f16(false, a, false, bm,
                                               (short)0, acc, false, false);
  #pragma unroll
  for (int v = 0; v < 8; ++v) {
    const int i = i0 + v + 8 * h;
    Th[((size_t)b * SEQ + i) * NCOL + col0 + n] = (_Float16)acc[v];
  }
}

// ---------------- K4: stage-3 GEMM: out[b,k] = T_bk (192x32) @ p2[b]^T (32x192)
// One wave per 16x16 output tile; +bias, /BATCH, f32 store.
__global__ void stage3_kernel(const _Float16* __restrict__ Th,
                              const _Float16* __restrict__ p2h,
                              const float* __restrict__ bout,
                              float* __restrict__ out) {
  const int wave = (int)((blockIdx.x * blockDim.x + threadIdx.x) >> 5);
  const int lane = threadIdx.x & 31;
  const int b   = wave / (CI * 144);
  const int rem = wave % (CI * 144);
  const int k = rem / 144;
  const int t = rem % 144;
  const int i0 = (t / 12) * 16, j0 = (t % 12) * 16;
  const int n = lane & 15, h = lane >> 4;

  // A: rows of T for this (b,k): T[b, i, k*32 + d]
  const _Float16* Arow = Th + ((size_t)b * SEQ + (i0 + n)) * NCOL + k * C;
  v16h a, bm;
  #pragma unroll
  for (int v = 0; v < 8; ++v) {
    const int kb = h * 16 + 2 * v;
    h2 pa = *(const h2*)(Arow + kb);
    a[2 * v] = pa.x; a[2 * v + 1] = pa.y;
  }
  // B: B[d][j] = p2[b, j0+n, d]; contiguous in d per lane
  const _Float16* Brow = p2h + ((size_t)b * SEQ + (j0 + n)) * C + h * 16;
  #pragma unroll
  for (int v = 0; v < 8; ++v) {
    h2 pb = *(const h2*)(Brow + 2 * v);
    bm[2 * v] = pb.x; bm[2 * v + 1] = pb.y;
  }

  v8f acc = {};
  acc = __builtin_amdgcn_wmma_f32_16x16x32_f16(false, a, false, bm,
                                               (short)0, acc, false, false);
  const float bk = bout[k];
  #pragma unroll
  for (int v = 0; v < 8; ++v) {
    const int i = i0 + v + 8 * h;
    out[(((size_t)b * CI + k) * SEQ + i) * SEQ + j0 + n] =
        (acc[v] + bk) * (1.0f / (float)BATCH);
  }
}

extern "C" void kernel_launch(void* const* d_in, const int* in_sizes, int n_in,
                              void* d_out, int out_size, void* d_ws, size_t ws_size,
                              hipStream_t stream) {
  const float* p     = (const float*)d_in[0];
  const float* gamma = (const float*)d_in[1];
  const float* beta  = (const float*)d_in[2];
  const float* W1    = (const float*)d_in[3];
  const float* b1    = (const float*)d_in[4];
  const float* W2    = (const float*)d_in[5];
  const float* b2    = (const float*)d_in[6];
  const float* Wout  = (const float*)d_in[7];
  const float* bout  = (const float*)d_in[8];
  float* out = (float*)d_out;

  _Float16* p1h = (_Float16*)d_ws;                 // 2*192*32
  _Float16* p2h = p1h + (size_t)BATCH * SEQ * C;   // 2*192*32
  _Float16* Whc = p2h + (size_t)BATCH * SEQ * C;   // 128*32*32
  _Float16* Th  = Whc + (size_t)CI * C * C;        // 2*192*4096

  // K1: LN + projections (384 rows)
  ln_proj_kernel<<<SEQ * BATCH, 128, 0, stream>>>(p, gamma, beta, W1, b1, W2, b2,
                                                  p1h, p2h);
  // K2: weight repack (131072 elems)
  wprep_kernel<<<(CI * C * C + 255) / 256, 256, 0, stream>>>(Wout, Whc);
  // K3: 2 * 12 * 256 = 6144 tiles, 8 waves/block -> 768 blocks
  stage2_kernel<<<768, 256, 0, stream>>>(p1h, Whc, Th);
  // K4: 2 * 128 * 144 = 36864 tiles, 8 waves/block -> 4608 blocks
  stage3_kernel<<<4608, 256, 0, stream>>>(Th, p2h, bout, out);
}
